// scaled_dotproduct_attention_37340445672065
// MI455X (gfx1250) — matrix operand order repaired and checked by reading, not compiled, not run on hardware
//
#include <hip/hip_runtime.h>
#include <hip/hip_bf16.h>

typedef _Float16 half_t;
typedef __attribute__((ext_vector_type(16))) _Float16 v16h;
typedef __attribute__((ext_vector_type(8)))  _Float16 v8h;
typedef __attribute__((ext_vector_type(8)))  float    v8f;

#define D_HEAD 64
#define SCALE  0.125f
#define NEG_INF -1e9f

#define WAVES 4
#define THREADS (WAVES * 32)
#define ROWS_PER_BLOCK 16
#define CHUNK 32   // keys per chunk (one f16 WMMA K-dim for the PV GEMM)

// ---------------- Prep: K -> f16 row-major, V -> f16 transposed [64, nkeys] ----------------
__global__ void prep_kernel(const float* __restrict__ K, const float* __restrict__ V,
                            half_t* __restrict__ Kh, half_t* __restrict__ Vt, int nkeys)
{
    const size_t nk64 = (size_t)nkeys * D_HEAD;
    for (size_t i = (size_t)blockIdx.x * blockDim.x + threadIdx.x; i < 2 * nk64;
         i += (size_t)gridDim.x * blockDim.x) {
        if (i < nk64) {
            Kh[i] = (half_t)K[i];
        } else {
            const size_t idx = i - nk64;
            const size_t k = idx % (size_t)nkeys;   // key index (fast axis -> coalesced write)
            const size_t n = idx / (size_t)nkeys;   // head-dim index
            Vt[idx] = (half_t)V[k * D_HEAD + n];
        }
    }
}

template <bool FAST>
__global__ __launch_bounds__(THREADS)
void attn_fused_kernel(const float* __restrict__ Q,
                       const float* __restrict__ K,
                       const float* __restrict__ V,
                       const half_t* __restrict__ Kh,   // f16 K, row-major (FAST path)
                       const half_t* __restrict__ Vt,   // f16 V^T [64, nkeys] (FAST path)
                       const unsigned char* __restrict__ mask,
                       float* __restrict__ ctx_out,     // [N, 64]
                       float* __restrict__ prob_out,    // [N, nkeys] (also S scratch)
                       int nkeys)
{
    __shared__ float red[WAVES][ROWS_PER_BLOCK];
    __shared__ __align__(16) half_t stage[WAVES][ROWS_PER_BLOCK * CHUNK];
    __shared__ float ctx_lds[ROWS_PER_BLOCK * D_HEAD];

    const int tid  = threadIdx.x;
    const int lane = tid & 31;
    const int wave = tid >> 5;
    const int rowbase = blockIdx.x * ROWS_PER_BLOCK;

    const int lm = lane & 15;   // column / M index within 16-group
    const int lh = lane >> 4;   // half select
    const int nchunk = nkeys / CHUNK;

    // ---- Load Q as f16 A-operands (k=0..31 and k=32..63)
    // A operand, this code's contiguous k order: lane -> M = lane&15 ; element e -> K = 16*lh + e
    v16h a0, a1;
    {
        const float* qrow = Q + (size_t)(rowbase + lm) * D_HEAD;
        const int kb = 16 * lh;
        #pragma unroll
        for (int e = 0; e < 16; ++e) {
            const int k = kb + e;
            a0[e] = (half_t)qrow[k];
            a1[e] = (half_t)qrow[k + 32];
        }
    }

    // ---- Phase 1: S = scale*QK^T masked; store to prob buffer; row max
    float pmax[8];
    #pragma unroll
    for (int r = 0; r < 8; ++r) pmax[r] = NEG_INF;

    for (int c = wave; c < nchunk; c += WAVES) {
        const int jb = c * CHUNK;
        // B operand, same contiguous k order as A: lane -> N = lane&15 ; element e -> K = 16*lh + e
        v16h b00, b01, b10, b11;
        if (FAST) {
            const half_t* k0 = Kh + (size_t)(jb + lm)      * D_HEAD + 16 * lh;
            const half_t* k1 = Kh + (size_t)(jb + 16 + lm) * D_HEAD + 16 * lh;
            b00 = *(const v16h*)(k0);
            b01 = *(const v16h*)(k0 + 32);
            b10 = *(const v16h*)(k1);
            b11 = *(const v16h*)(k1 + 32);
        } else {
            const float* krow0 = K + (size_t)(jb + lm)      * D_HEAD + 16 * lh;
            const float* krow1 = K + (size_t)(jb + 16 + lm) * D_HEAD + 16 * lh;
            #pragma unroll
            for (int e = 0; e < 16; ++e) {
                b00[e] = (half_t)krow0[e];
                b01[e] = (half_t)krow0[e + 32];
                b10[e] = (half_t)krow1[e];
                b11[e] = (half_t)krow1[e + 32];
            }
        }
        v8f s0 = {}; v8f s1 = {};
        s0 = __builtin_amdgcn_wmma_f32_16x16x32_f16(false, a0, false, b00, (short)0, s0, false, false);
        s0 = __builtin_amdgcn_wmma_f32_16x16x32_f16(false, a1, false, b01, (short)0, s0, false, false);
        s1 = __builtin_amdgcn_wmma_f32_16x16x32_f16(false, a0, false, b10, (short)0, s1, false, false);
        s1 = __builtin_amdgcn_wmma_f32_16x16x32_f16(false, a1, false, b11, (short)0, s1, false, false);

        // C layout: vgpr r -> row M = r + 8*lh, col = lm
        #pragma unroll
        for (int r = 0; r < 8; ++r) {
            const size_t base = (size_t)(rowbase + r + 8 * lh) * (size_t)nkeys + jb + lm;
            const float v0 = mask[base]      ? NEG_INF : s0[r] * SCALE;
            const float v1 = mask[base + 16] ? NEG_INF : s1[r] * SCALE;
            prob_out[base]      = v0;   // stash masked scores (L2-resident scratch)
            prob_out[base + 16] = v1;
            const float m = v0 > v1 ? v0 : v1;
            pmax[r] = pmax[r] > m ? pmax[r] : m;
        }
    }

    // reduce max across the 16 lanes of each half (xor stays within half)
    #pragma unroll
    for (int r = 0; r < 8; ++r) {
        #pragma unroll
        for (int off = 1; off < 16; off <<= 1) {
            const float o = __shfl_xor(pmax[r], off, 32);
            pmax[r] = pmax[r] > o ? pmax[r] : o;
        }
    }
    if (lm == 0) {
        #pragma unroll
        for (int r = 0; r < 8; ++r) red[wave][r + 8 * lh] = pmax[r];
    }
    __syncthreads();
    float rowmax[8];
    #pragma unroll
    for (int r = 0; r < 8; ++r) {
        float m = red[0][r + 8 * lh];
        #pragma unroll
        for (int w = 1; w < WAVES; ++w) { const float o = red[w][r + 8 * lh]; m = m > o ? m : o; }
        rowmax[r] = m;
    }
    __syncthreads();   // red[] reused for sums

    // ---- Phase 2: row sums of exp(s - max)
    float psum[8];
    #pragma unroll
    for (int r = 0; r < 8; ++r) psum[r] = 0.f;

    for (int c = wave; c < nchunk; c += WAVES) {
        const int jb = c * CHUNK;
        #pragma unroll
        for (int r = 0; r < 8; ++r) {
            const size_t base = (size_t)(rowbase + r + 8 * lh) * (size_t)nkeys + jb + lm;
            psum[r] += __expf(prob_out[base]      - rowmax[r]);
            psum[r] += __expf(prob_out[base + 16] - rowmax[r]);
        }
    }
    #pragma unroll
    for (int r = 0; r < 8; ++r) {
        #pragma unroll
        for (int off = 1; off < 16; off <<= 1)
            psum[r] += __shfl_xor(psum[r], off, 32);
    }
    if (lm == 0) {
        #pragma unroll
        for (int r = 0; r < 8; ++r) red[wave][r + 8 * lh] = psum[r];
    }
    __syncthreads();
    float rinv[8];
    #pragma unroll
    for (int r = 0; r < 8; ++r) {
        float s = 0.f;
        #pragma unroll
        for (int w = 0; w < WAVES; ++w) s += red[w][r + 8 * lh];
        rinv[r] = 1.0f / s;   // sum >= 1 always (max element contributes exp(0))
    }

    // ---- Phase 3: write normalized P, context += P @ V
    v8f ctx[4] = {};
    for (int c = wave; c < nchunk; c += WAVES) {
        const int jb = c * CHUNK;
        #pragma unroll
        for (int r = 0; r < 8; ++r) {
            const int M = r + 8 * lh;
            const size_t base = (size_t)(rowbase + M) * (size_t)nkeys + jb + lm;
            const float p0 = __expf(prob_out[base]      - rowmax[r]) * rinv[r];
            const float p1 = __expf(prob_out[base + 16] - rowmax[r]) * rinv[r];
            prob_out[base]      = p0;   // final output value
            prob_out[base + 16] = p1;
            stage[wave][M * CHUNK + lm]      = (half_t)p0;  // C-layout -> LDS
            stage[wave][M * CHUNK + 16 + lm] = (half_t)p1;
        }
        // read back P in A layout: row M = lm, K = 16*lh + e
        v16h pa;
        {
            const half_t* srow = &stage[wave][lm * CHUNK + 16 * lh];
            const v8h lo = *(const v8h*)(srow);
            const v8h hi = *(const v8h*)(srow + 8);
            #pragma unroll
            for (int e = 0; e < 8; ++e) { pa[e] = lo[e]; pa[e + 8] = hi[e]; }
        }
        // V in B layout per 16-wide n-tile: N = lm, key K = jb + 16*lh + e
        #pragma unroll
        for (int nt = 0; nt < 4; ++nt) {
            v16h bv;
            if (FAST) {
                // Vt[n][k] contiguous in k: one 32B vector load per operand
                bv = *(const v16h*)(Vt + (size_t)(nt * 16 + lm) * (size_t)nkeys + jb + 16 * lh);
            } else {
                const float* vcol = V + (size_t)(jb + 16 * lh) * D_HEAD + nt * 16 + lm;
                #pragma unroll
                for (int e = 0; e < 16; ++e)
                    bv[e] = (half_t)vcol[(size_t)e * D_HEAD];
            }
            ctx[nt] = __builtin_amdgcn_wmma_f32_16x16x32_f16(false, pa, false, bv, (short)0, ctx[nt], false, false);
        }
    }

    // cross-wave context reduction via LDS float atomics
    for (int i = tid; i < ROWS_PER_BLOCK * D_HEAD; i += THREADS) ctx_lds[i] = 0.f;
    __syncthreads();
    #pragma unroll
    for (int nt = 0; nt < 4; ++nt) {
        #pragma unroll
        for (int r = 0; r < 8; ++r) {
            const int M = r + 8 * lh;
            atomicAdd(&ctx_lds[M * D_HEAD + nt * 16 + lm], ctx[nt][r]);
        }
    }
    __syncthreads();
    for (int i = tid; i < ROWS_PER_BLOCK * D_HEAD; i += THREADS) {
        const int M = i / D_HEAD, n = i % D_HEAD;
        ctx_out[(size_t)(rowbase + M) * D_HEAD + n] = ctx_lds[i];
    }
}

extern "C" void kernel_launch(void* const* d_in, const int* in_sizes, int n_in,
                              void* d_out, int out_size, void* d_ws, size_t ws_size,
                              hipStream_t stream) {
    const float* Q = (const float*)d_in[0];
    const float* K = (const float*)d_in[1];
    const float* V = (const float*)d_in[2];
    const unsigned char* mask = (const unsigned char*)d_in[3];

    const int N     = in_sizes[0] / D_HEAD;   // 8192 query rows
    const int nkeys = in_sizes[1] / D_HEAD;   // 8192 keys

    float* ctx_out  = (float*)d_out;                       // [N, 64] first (tuple order)
    float* prob_out = (float*)d_out + (size_t)N * D_HEAD;  // [N, nkeys]

    const size_t nk64 = (size_t)nkeys * D_HEAD;
    const size_t need = 2 * nk64 * sizeof(half_t);         // Kh + Vt
    const int grid = N / ROWS_PER_BLOCK;

    if (ws_size >= need) {
        half_t* Kh = (half_t*)d_ws;
        half_t* Vt = Kh + nk64;
        prep_kernel<<<2048, 256, 0, stream>>>(K, V, Kh, Vt, nkeys);
        attn_fused_kernel<true><<<grid, THREADS, 0, stream>>>(
            Q, K, V, Kh, Vt, mask, ctx_out, prob_out, nkeys);
    } else {
        attn_fused_kernel<false><<<grid, THREADS, 0, stream>>>(
            Q, K, V, (const half_t*)nullptr, (const half_t*)nullptr,
            mask, ctx_out, prob_out, nkeys);
    }
}
